// SPINCS_86406152061260
// MI455X (gfx1250) — matrix operand order repaired and checked by reading, not compiled, not run on hardware
//
#include <hip/hip_runtime.h>
#include <hip/hip_bf16.h>

// ---------------- problem constants (from reference) ----------------
#define NN   50000            // nodes
#define EE   800000           // edges (before self loops)
#define ETOT (EE + NN)        // edges incl. self loops
#define FD   128              // feature dim = HEADS*HID = IN
#define NH   4                // heads
#define HC   32               // hidden per head
#define NG   64               // graphs
#define NO   10               // classes
#define NEG_SLOPE 0.2f

typedef __attribute__((ext_vector_type(16))) _Float16 v16h;
typedef __attribute__((ext_vector_type(8)))  _Float16 v8h;
typedef __attribute__((ext_vector_type(8)))  float    v8f;

// ---------------- utility kernels ----------------
__global__ void fill_f32(float* __restrict__ p, float v, int n) {
    int i = blockIdx.x * blockDim.x + threadIdx.x;
    if (i < n) p[i] = v;
}

// f32 -> f16 straight copy (features)
__global__ void cvt_f16(const float* __restrict__ src, _Float16* __restrict__ dst, int n) {
    int i = blockIdx.x * blockDim.x + threadIdx.x;
    if (i < n) dst[i] = (_Float16)src[i];
}

// W [K=128][N=128] f32 (row-major)  ->  Wt [N][K] f16 (transposed)
__global__ void cvt_w_transpose(const float* __restrict__ W, _Float16* __restrict__ Wt) {
    int i = blockIdx.x * blockDim.x + threadIdx.x;   // i < 128*128
    if (i >= FD * FD) return;
    int k = i / FD, n = i % FD;
    Wt[n * FD + k] = (_Float16)W[k * FD + n];
}

// ---------------- WMMA GEMM: Out[N,128] = A16[N,128] @ W (via Wt[N][K] f16) ----------------
// One wave computes a full 16x128 row-block: 8 accumulators (16x16 tiles), K=128 in 4 steps.
// A fragment is loaded ONCE per K-step and reused across all 8 column tiles (32 WMMAs/wave).
// A-fragment layout (16-bit A 16x32, contiguous K as in B): lanes 0-15 row M=l, K = k..k+15;
// lanes 16-31 row M=l-16, K = k+16..k+31.
// B-fragment (32x16): lanes 0-15 col N=l, K = k..k+15; lanes 16-31 col N=l-16, K = k+16..k+31.
// Block = 160 threads = 5 waves; 3125 row tiles / 5 = 625 blocks exactly (EXEC always all-ones).
__global__ void gemm_wmma_f16(const _Float16* __restrict__ A,
                              const _Float16* __restrict__ Bt,   // [N=128][K=128]
                              float* __restrict__ Out) {
    const int wave    = threadIdx.x >> 5;
    const int lane    = threadIdx.x & 31;
    const int rowBase = (blockIdx.x * 5 + wave) * 16;
    const bool hi     = lane >= 16;
    const int  l16    = lane & 15;

    const _Float16* ap = A  + (size_t)(rowBase + l16) * FD;
    const _Float16* bp = Bt + (size_t)l16 * FD;        // + ct*16*FD per column tile

    v8f acc[8];
#pragma unroll
    for (int ct = 0; ct < 8; ++ct) acc[ct] = (v8f){};

#pragma unroll
    for (int k = 0; k < FD; k += 32) {
        const int ka = k + (hi ? 16 : 0);
        v8h alo = *(const v8h*)(ap + ka);              // K = ka..ka+7      (16B aligned)
        v8h ahi = *(const v8h*)(ap + ka + 8);          // K = ka+8..ka+15
        v16h a;
#pragma unroll
        for (int i = 0; i < 8; ++i) { a[i] = alo[i]; a[8 + i] = ahi[i]; }

        const int kb = k + (hi ? 16 : 0);
#pragma unroll
        for (int ct = 0; ct < 8; ++ct) {
            v16h b = *(const v16h*)(bp + ct * (16 * FD) + kb);   // 16 contiguous halves
            acc[ct] = __builtin_amdgcn_wmma_f32_16x16x32_f16(
                /*neg_a=*/false, a, /*neg_b=*/false, b,
                /*c_mod=*/(short)0, acc[ct], /*reuse_a=*/false, /*reuse_b=*/false);
        }
    }

    // C/D layout: lanes 0-15: N=l, VGPR r -> M=r ; lanes 16-31: N=l-16, VGPR r -> M=8+r
    const int rb = rowBase + (hi ? 8 : 0);
#pragma unroll
    for (int ct = 0; ct < 8; ++ct) {
        const int col = ct * 16 + l16;
#pragma unroll
        for (int r = 0; r < 8; ++r) Out[(size_t)(rb + r) * FD + col] = acc[ct][r];
    }
}

// ---------------- attention logits per node ----------------
// als[n][h] = dot(t[n, h*32 : h*32+32], a_src[h]) ; ald analogous
__global__ void node_attn(const float* __restrict__ t,
                          const float* __restrict__ a_src,
                          const float* __restrict__ a_dst,
                          float* __restrict__ als, float* __restrict__ ald) {
    int n = blockIdx.x * blockDim.x + threadIdx.x;
    if (n >= NN) return;
    const float* row = t + (size_t)n * FD;
#pragma unroll
    for (int h = 0; h < NH; ++h) {
        float s = 0.f, d = 0.f;
#pragma unroll 8
        for (int c = 0; c < HC; ++c) {
            float v = row[h * HC + c];
            s += v * a_src[h * HC + c];
            d += v * a_dst[h * HC + c];
        }
        als[n * NH + h] = s;
        ald[n * NH + h] = d;
    }
}

// ---------------- edge helpers ----------------
__device__ __forceinline__ void edge_ep(const int* __restrict__ ei, int e, int& s, int& d) {
    if (e < EE) { s = ei[e]; d = ei[EE + e]; }
    else        { s = d = e - EE; }               // self loops
}
__device__ __forceinline__ float leaky(float v) { return v > 0.f ? v : NEG_SLOPE * v; }

// monotone float max via int/uint atomics (no NaNs present; -inf init)
__device__ __forceinline__ void atomicMaxF(float* addr, float v) {
    if (v >= 0.f) atomicMax((int*)addr, __float_as_int(v));
    else          atomicMin((unsigned int*)addr, __float_as_uint(v));
}

// pass 1: segment max of leaky(als[src]+ald[dst]) into m[dst][h]
__global__ void edge_max(const int* __restrict__ ei,
                         const float* __restrict__ als, const float* __restrict__ ald,
                         float* __restrict__ m) {
    int idx = blockIdx.x * blockDim.x + threadIdx.x;
    if (idx >= ETOT * NH) return;
    int e = idx >> 2, h = idx & 3, s, d;
    edge_ep(ei, e, s, d);
    atomicMaxF(&m[d * NH + h], leaky(als[s * NH + h] + ald[d * NH + h]));
}

// pass 2: z[dst][h] += exp(e - m[dst][h])
__global__ void edge_sum(const int* __restrict__ ei,
                         const float* __restrict__ als, const float* __restrict__ ald,
                         const float* __restrict__ m, float* __restrict__ z) {
    int idx = blockIdx.x * blockDim.x + threadIdx.x;
    if (idx >= ETOT * NH) return;
    int e = idx >> 2, h = idx & 3, s, d;
    edge_ep(ei, e, s, d);
    float v = leaky(als[s * NH + h] + ald[d * NH + h]);
    atomicAdd(&z[d * NH + h], __expf(v - m[d * NH + h]));
}

// pass 3: agg[dst] += h[src] * alpha ; one wave per edge, lane handles col c=lane in each head
__global__ void edge_agg(const int* __restrict__ ei,
                         const float* __restrict__ als, const float* __restrict__ ald,
                         const float* __restrict__ m, const float* __restrict__ z,
                         const float* __restrict__ t, float* __restrict__ agg) {
    int e    = blockIdx.x * (blockDim.x >> 5) + (threadIdx.x >> 5);
    int lane = threadIdx.x & 31;
    if (e >= ETOT) return;
    int s, d;
    edge_ep(ei, e, s, d);
    float alpha[NH];
#pragma unroll
    for (int h = 0; h < NH; ++h) {
        float v = leaky(als[s * NH + h] + ald[d * NH + h]);
        float p = __expf(v - m[d * NH + h]);
        alpha[h] = p / (z[d * NH + h] + 1e-16f);
    }
#pragma unroll
    for (int h = 0; h < NH; ++h) {
        int f = h * HC + lane;
        atomicAdd(&agg[(size_t)d * FD + f], t[(size_t)s * FD + f] * alpha[h]);
    }
}

// layer-1 epilogue: feat16 = f16(elu(agg + b1))   (feeds layer-2 WMMA GEMM directly)
__global__ void post_l1(const float* __restrict__ agg, const float* __restrict__ b1,
                        _Float16* __restrict__ feat16) {
    int i = blockIdx.x * blockDim.x + threadIdx.x;
    if (i >= NN * FD) return;
    float v = agg[i] + b1[i % FD];
    v = v > 0.f ? v : (__expf(v) - 1.f);          // ELU, alpha=1
    feat16[i] = (_Float16)v;
}

// mean-pool scatter: pooled[batch[n]] += agg[n] + b2 ; counts[batch[n]] += 1
__global__ void pool_scatter(const float* __restrict__ agg, const float* __restrict__ b2,
                             const int* __restrict__ batch,
                             float* __restrict__ pooled, float* __restrict__ counts) {
    int i = blockIdx.x * blockDim.x + threadIdx.x;
    if (i >= NN * FD) return;
    int n = i / FD, f = i % FD;
    int g = batch[n];
    atomicAdd(&pooled[g * FD + f], agg[i] + b2[f]);
    if (f == 0) atomicAdd(&counts[g], 1.0f);
}

// head: logits = (pooled/count) @ Wl + bl ; softmax rows ; one thread per graph
__global__ void head_softmax(const float* __restrict__ pooled, const float* __restrict__ counts,
                             const float* __restrict__ Wl, const float* __restrict__ bl,
                             float* __restrict__ out) {
    int g = threadIdx.x;
    if (g >= NG) return;
    float inv = 1.0f / fmaxf(counts[g], 1.0f);
    float logit[NO];
#pragma unroll
    for (int o = 0; o < NO; ++o) logit[o] = bl[o];
    for (int f = 0; f < FD; ++f) {
        float pv = pooled[g * FD + f] * inv;
#pragma unroll
        for (int o = 0; o < NO; ++o) logit[o] += pv * Wl[f * NO + o];
    }
    float mx = logit[0];
#pragma unroll
    for (int o = 1; o < NO; ++o) mx = fmaxf(mx, logit[o]);
    float ssum = 0.f;
#pragma unroll
    for (int o = 0; o < NO; ++o) { logit[o] = __expf(logit[o] - mx); ssum += logit[o]; }
    float rs = 1.0f / ssum;
#pragma unroll
    for (int o = 0; o < NO; ++o) out[g * NO + o] = logit[o] * rs;
}

// ---------------- host orchestration ----------------
static inline int cdiv(int a, int b) { return (a + b - 1) / b; }

extern "C" void kernel_launch(void* const* d_in, const int* in_sizes, int n_in,
                              void* d_out, int out_size, void* d_ws, size_t ws_size,
                              hipStream_t stream) {
    (void)in_sizes; (void)n_in; (void)out_size; (void)ws_size;

    const float* x    = (const float*)d_in[0];
    const float* W1   = (const float*)d_in[1];
    const float* a1s  = (const float*)d_in[2];
    const float* a1d  = (const float*)d_in[3];
    const float* b1   = (const float*)d_in[4];
    const float* W2   = (const float*)d_in[5];
    const float* a2s  = (const float*)d_in[6];
    const float* a2d  = (const float*)d_in[7];
    const float* b2   = (const float*)d_in[8];
    const float* Wl   = (const float*)d_in[9];
    const float* bl   = (const float*)d_in[10];
    const int*   ei   = (const int*)d_in[11];   // [2, E]
    const int*   bat  = (const int*)d_in[12];   // [N]
    float*       out  = (float*)d_out;          // [G, OUT]

    // workspace layout (all 256B aligned by construction)
    char* ws = (char*)d_ws;
    size_t off = 0;
    float*    t      = (float*)(ws + off);    off += (size_t)NN * FD * 4;   // 25.6 MB
    float*    agg    = (float*)(ws + off);    off += (size_t)NN * FD * 4;   // 25.6 MB
    _Float16* feat16 = (_Float16*)(ws + off); off += (size_t)NN * FD * 2;   // 12.8 MB
    _Float16* wt16   = (_Float16*)(ws + off); off += (size_t)FD * FD * 2;   // 32 KB
    float*    als    = (float*)(ws + off);    off += (size_t)NN * NH * 4;
    float*    ald    = (float*)(ws + off);    off += (size_t)NN * NH * 4;
    float*    m      = (float*)(ws + off);    off += (size_t)NN * NH * 4;
    float*    z      = (float*)(ws + off);    off += (size_t)NN * NH * 4;
    float*    pooled = (float*)(ws + off);    off += (size_t)NG * FD * 4;
    float*    counts = (float*)(ws + off);    off += (size_t)NG * 4;

    const int NF  = NN * FD;                   // 6.4M
    const int NEH = ETOT * NH;                 // 3.4M
    const int gemmBlocks = (NN / 16) / 5;      // 3125 row tiles, 5 waves/block -> 625 blocks
    const float NEG_INF = -__builtin_inff();

    // ---------- layer 1 ----------
    cvt_f16        <<<cdiv(NF, 256), 256, 0, stream>>>(x, feat16, NF);
    cvt_w_transpose<<<cdiv(FD * FD, 256), 256, 0, stream>>>(W1, wt16);
    gemm_wmma_f16  <<<gemmBlocks, 160, 0, stream>>>(feat16, wt16, t);

    fill_f32<<<cdiv(NF, 256), 256, 0, stream>>>(agg, 0.f, NF);
    fill_f32<<<cdiv(NN * NH, 256), 256, 0, stream>>>(m, NEG_INF, NN * NH);
    fill_f32<<<cdiv(NN * NH, 256), 256, 0, stream>>>(z, 0.f, NN * NH);

    node_attn<<<cdiv(NN, 128), 128, 0, stream>>>(t, a1s, a1d, als, ald);
    edge_max <<<cdiv(NEH, 256), 256, 0, stream>>>(ei, als, ald, m);
    edge_sum <<<cdiv(NEH, 256), 256, 0, stream>>>(ei, als, ald, m, z);
    edge_agg <<<cdiv(ETOT, 8), 256, 0, stream>>>(ei, als, ald, m, z, t, agg);

    post_l1<<<cdiv(NF, 256), 256, 0, stream>>>(agg, b1, feat16);

    // ---------- layer 2 ----------
    cvt_w_transpose<<<cdiv(FD * FD, 256), 256, 0, stream>>>(W2, wt16);
    gemm_wmma_f16  <<<gemmBlocks, 160, 0, stream>>>(feat16, wt16, t);

    fill_f32<<<cdiv(NF, 256), 256, 0, stream>>>(agg, 0.f, NF);
    fill_f32<<<cdiv(NN * NH, 256), 256, 0, stream>>>(m, NEG_INF, NN * NH);
    fill_f32<<<cdiv(NN * NH, 256), 256, 0, stream>>>(z, 0.f, NN * NH);

    node_attn<<<cdiv(NN, 128), 128, 0, stream>>>(t, a2s, a2d, als, ald);
    edge_max <<<cdiv(NEH, 256), 256, 0, stream>>>(ei, als, ald, m);
    edge_sum <<<cdiv(NEH, 256), 256, 0, stream>>>(ei, als, ald, m, z);
    edge_agg <<<cdiv(ETOT, 8), 256, 0, stream>>>(ei, als, ald, m, z, t, agg);

    // ---------- pool + head ----------
    fill_f32<<<cdiv(NG * FD, 256), 256, 0, stream>>>(pooled, 0.f, NG * FD);
    fill_f32<<<1, 64, 0, stream>>>(counts, 0.f, NG);
    pool_scatter<<<cdiv(NF, 256), 256, 0, stream>>>(agg, b2, bat, pooled, counts);
    head_softmax<<<1, 64, 0, stream>>>(pooled, counts, Wl, bl, out);
}
